// LocalAttention_5162550690368
// MI455X (gfx1250) — matrix operand order repaired and checked by reading, not compiled, not run on hardware
//
#include <hip/hip_runtime.h>

// ---------- CDNA5 WMMA types ----------
typedef __attribute__((ext_vector_type(16))) __bf16 v16bf;
typedef __attribute__((ext_vector_type(8)))  float  v8f;

#define WIN    128
#define JW     384      // 3*WIN look-around keys
#define HEADS  8
#define DHEAD  64
#define NSEQ   4096
#define DMODEL 512
#define NWIN   32       // NSEQ / WIN
#define KSTR   72       // Q/K LDS row stride (ushorts): 144 B, 16-B aligned, conflict-free
#define VSTR   392      // V^T LDS row stride (ushorts): 784 B, 16-B aligned, 4-bank skew
#define PSTR   40       // P staging row stride (ushorts): 80 B, 16-B aligned

union FragBF {
    v16bf v;
    uint4 q[2];
    unsigned int  w[8];
    unsigned short u[16];
};

__device__ __forceinline__ unsigned short f2bf(float x) {           // RNE
    unsigned int u = __float_as_uint(x);
    u += 0x7FFFu + ((u >> 16) & 1u);
    return (unsigned short)(u >> 16);
}
__device__ __forceinline__ unsigned int f2bf_pk(float lo, float hi) {
    return (unsigned int)f2bf(lo) | ((unsigned int)f2bf(hi) << 16);
}
__device__ __forceinline__ float fast_exp2(float x) {
#if __has_builtin(__builtin_amdgcn_exp2f)
    return __builtin_amdgcn_exp2f(x);     // raw v_exp_f32 (flushes subnormal result)
#else
    return exp2f(x);
#endif
}

__global__ __launch_bounds__(256) void local_attn_wmma(
    const float* __restrict__ q, const float* __restrict__ k,
    const float* __restrict__ v, const unsigned char* __restrict__ mask,
    float* __restrict__ out)
{
    extern __shared__ unsigned short smem[];
    unsigned short* lds_q  = smem;                        // 128*72
    unsigned short* lds_k  = lds_q + 128 * KSTR;          // 384*72
    unsigned short* lds_vt = lds_k + JW  * KSTR;          // 64*392 (V transposed: [c][j])
    unsigned short* lds_p  = lds_vt + DHEAD * VSTR;       // 8 waves * 16*40
    float*          lds_bias = (float*)(lds_p + 8 * 16 * PSTR);  // 384 f32 mask biases

    const int wi = blockIdx.x & (NWIN - 1);
    const int bh = blockIdx.x >> 5;
    const int bb = bh >> 3;            // / HEADS
    const int hh = bh & 7;

    const int tid  = threadIdx.x;
    const int lane = tid & 31;
    const int wave = tid >> 5;
    const int hi   = lane >> 4;        // half-wave select
    const int ln   = lane & 15;

    // ---------------- Phase 1: global -> LDS (f32 -> bf16) ----------------
    const float* qbase = q + ((size_t)bb * NSEQ + (size_t)wi * WIN) * DMODEL + hh * DHEAD;
    for (int idx = tid; idx < 128 * 16; idx += 256) {
        int row = idx >> 4, c4 = (idx & 15) << 2;
        float4 f = *(const float4*)(qbase + (size_t)row * DMODEL + c4);
        unsigned int* d = (unsigned int*)(lds_q + row * KSTR + c4);
        d[0] = f2bf_pk(f.x, f.y);
        d[1] = f2bf_pk(f.z, f.w);
    }
    const unsigned short NEG1 = 0xBF80;  // bf16(-1.0): reference pads k/v values with -1
    for (int idx = tid; idx < JW * 16; idx += 256) {
        int row = idx >> 4, c4 = (idx & 15) << 2;        // row = key j, c4 = feature col
        int tok = wi * WIN - WIN + row;
        unsigned int* dk = (unsigned int*)(lds_k + row * KSTR + c4);
        unsigned short* dv = lds_vt + c4 * VSTR + row;   // transposed store
        if (tok >= 0 && tok < NSEQ) {
            size_t off = ((size_t)bb * NSEQ + tok) * DMODEL + hh * DHEAD + c4;
            float4 fk = *(const float4*)(k + off);
            float4 fv = *(const float4*)(v + off);
            dk[0] = f2bf_pk(fk.x, fk.y);
            dk[1] = f2bf_pk(fk.z, fk.w);
            dv[0 * VSTR] = f2bf(fv.x); dv[1 * VSTR] = f2bf(fv.y);
            dv[2 * VSTR] = f2bf(fv.z); dv[3 * VSTR] = f2bf(fv.w);
        } else {
            dk[0] = dk[1] = (unsigned int)NEG1 | ((unsigned int)NEG1 << 16);
            dv[0 * VSTR] = NEG1; dv[1 * VSTR] = NEG1;
            dv[2 * VSTR] = NEG1; dv[3 * VSTR] = NEG1;
        }
    }
    // look-around mask -> additive score bias: 0 (attend) or -1e30 (exp2 -> 0)
    // clamped address => unconditional load, no exec divergence
    {
        const unsigned char* mbase = mask + (size_t)bb * NSEQ;
        for (int j = tid; j < JW; j += 256) {
            int tok  = wi * WIN - WIN + j;
            int tokc = tok < 0 ? 0 : (tok > NSEQ - 1 ? NSEQ - 1 : tok);
            unsigned char m = mbase[tokc];
            bool ok = (tok >= 0) && (tok < NSEQ) && (m != 0);
            lds_bias[j] = ok ? 0.0f : -1.0e30f;
        }
    }
    __syncthreads();

    // ---------------- Phase 2: per-wave 16-row strip -----------------------
    const int ibase = wave * 16;

    // Q A-fragments (16x32 bf16), two K-steps covering d=64.
    // A-layout: elements 0..15 hold K = 16*hi..16*hi+15 (contiguous)
    FragBF aQ0, aQ1;
    {
        const unsigned short* qr = lds_q + (ibase + ln) * KSTR + 16 * hi;
        aQ0.q[0] = *(const uint4*)(qr);
        aQ0.q[1] = *(const uint4*)(qr + 8);
        aQ1.q[0] = *(const uint4*)(qr + 32);
        aQ1.q[1] = *(const uint4*)(qr + 40);
    }

    // all-ones bf16 B fragment: row-sum of P via WMMA (replaces lane shuffles)
    FragBF ones;
    #pragma unroll
    for (int i = 0; i < 8; ++i) ones.w[i] = 0x3F803F80u;

    v8f oacc[4];
    v8f ssum = {};
    {
        v8f z = {};
        #pragma unroll
        for (int t = 0; t < 4; ++t) oacc[t] = z;
    }

    // exp(s*scale) == exp2(s * scale * log2(e) + bias); scores are O(1), no
    // max-shift needed for fp32 exp (overflow at ~88); bias=-1e30 kills masked cols.
    const float SCL2 = 0.044194173824159216f * 1.44269504088896340736f;
    unsigned short* pst = lds_p + wave * (16 * PSTR);

    #pragma unroll 4
    for (int jc = 0; jc < 12; ++jc) {
        const int jb = jc * 32;

        // ---- S chunk: two 16-col tiles, K=64 in two WMMA steps each ----
        v8f s0 = {}, s1 = {};
        {
            // B-layout: elements 0..15 hold K = 16*hi..16*hi+15 (contiguous)
            const unsigned short* kr0 = lds_k + (jb + ln) * KSTR + 16 * hi;
            const unsigned short* kr1 = kr0 + 16 * KSTR;
            FragBF b00, b01, b10, b11;
            b00.q[0] = *(const uint4*)(kr0);      b00.q[1] = *(const uint4*)(kr0 + 8);
            b01.q[0] = *(const uint4*)(kr0 + 32); b01.q[1] = *(const uint4*)(kr0 + 40);
            b10.q[0] = *(const uint4*)(kr1);      b10.q[1] = *(const uint4*)(kr1 + 8);
            b11.q[0] = *(const uint4*)(kr1 + 32); b11.q[1] = *(const uint4*)(kr1 + 40);
            s0 = __builtin_amdgcn_wmma_f32_16x16x32_bf16(false, aQ0.v, false, b00.v, (short)0, s0, false, false);
            s0 = __builtin_amdgcn_wmma_f32_16x16x32_bf16(false, aQ1.v, false, b01.v, (short)0, s0, false, false);
            s1 = __builtin_amdgcn_wmma_f32_16x16x32_bf16(false, aQ0.v, false, b10.v, (short)0, s1, false, false);
            s1 = __builtin_amdgcn_wmma_f32_16x16x32_bf16(false, aQ1.v, false, b11.v, (short)0, s1, false, false);
        }

        // ---- branch-free masked exp: 1 FMA + 1 v_exp per element ----
        const float b0 = lds_bias[jb + ln];
        const float b1 = lds_bias[jb + ln + 16];

        #pragma unroll
        for (int r = 0; r < 8; ++r) {
            float e0 = fast_exp2(fmaf(s0[r], SCL2, b0));
            float e1 = fast_exp2(fmaf(s1[r], SCL2, b1));
            int row = r + 8 * hi;
            // truncating f32->bf16 (P >= 0, no max-shift; trunc error <= bf16 ulp)
            pst[row * PSTR + ln]      = (unsigned short)(__float_as_uint(e0) >> 16);
            pst[row * PSTR + ln + 16] = (unsigned short)(__float_as_uint(e1) >> 16);
        }

        // ---- read P back in A-layout (two b128 loads) ----
        FragBF aP;
        {
            const unsigned short* pr = pst + ln * PSTR + 16 * hi;
            aP.q[0] = *(const uint4*)(pr);
            aP.q[1] = *(const uint4*)(pr + 8);
        }

        // ---- row sums: ssum += P * ones  (C-layout == per-row registers) ----
        ssum = __builtin_amdgcn_wmma_f32_16x16x32_bf16(false, aP.v, false, ones.v, (short)0, ssum, false, false);

        // ---- O += P * V : batch all 8 b128 loads, then 4 WMMAs ----
        FragBF bV[4];
        #pragma unroll
        for (int t = 0; t < 4; ++t) {
            const unsigned short* vr = lds_vt + (t * 16 + ln) * VSTR + jb + 16 * hi;
            bV[t].q[0] = *(const uint4*)(vr);
            bV[t].q[1] = *(const uint4*)(vr + 8);
        }
        #pragma unroll
        for (int t = 0; t < 4; ++t)
            oacc[t] = __builtin_amdgcn_wmma_f32_16x16x32_bf16(false, aP.v, false, bV[t].v, (short)0, oacc[t], false, false);
    }

    // ---------------- Epilogue: normalize + store f32 ----------------
    float* obase = out + ((size_t)bb * NSEQ + (size_t)wi * WIN + ibase) * DMODEL + hh * DHEAD;
    #pragma unroll
    for (int r = 0; r < 8; ++r) {
        float inv = ssum[r] > 0.0f ? 1.0f / ssum[r] : 0.0f;
        int row = r + 8 * hi;
        #pragma unroll
        for (int t = 0; t < 4; ++t)
            obase[(size_t)row * DMODEL + t * 16 + ln] = oacc[t][r] * inv;
    }
}

extern "C" void kernel_launch(void* const* d_in, const int* in_sizes, int n_in,
                              void* d_out, int out_size, void* d_ws, size_t ws_size,
                              hipStream_t stream) {
    const float* q = (const float*)d_in[0];
    const float* k = (const float*)d_in[1];
    const float* v = (const float*)d_in[2];
    const unsigned char* mask = (const unsigned char*)d_in[3];  // jnp bool -> 1 byte
    float* out = (float*)d_out;

    const size_t shmem = (size_t)(128 * KSTR + JW * KSTR + DHEAD * VSTR + 8 * 16 * PSTR)
                         * sizeof(unsigned short) + JW * sizeof(float);   // +1536 B bias
    hipFuncSetAttribute((const void*)local_attn_wmma,
                        hipFuncAttributeMaxDynamicSharedMemorySize, (int)shmem);

    const int nblocks = 4 * HEADS * NWIN;  // 1024 tiles
    local_attn_wmma<<<nblocks, 256, shmem, stream>>>(q, k, v, mask, out);
}
